// AttentionDynamicModel_20160576487798
// MI455X (gfx1250) — matrix operand order repaired and checked by reading, compiled, not run on hardware
//
#include <hip/hip_runtime.h>

// Problem constants (match reference)
#define TT 128
#define BB 256
#define NN 512
#define DD 512
#define HH 8
#define NEGV (-1.0e9f)

typedef __bf16 bf16x16 __attribute__((ext_vector_type(16)));
typedef __bf16 bf16x8  __attribute__((ext_vector_type(8)));
typedef float  f32x8   __attribute__((ext_vector_type(8)));

union FragU { bf16x16 v; bf16x8 h[2]; unsigned int u[8]; };

__device__ __forceinline__ unsigned short f2bf(float x) {
  unsigned int u = __float_as_uint(x);
  u += 0x7FFFu + ((u >> 16) & 1u);   // round-to-nearest-even
  return (unsigned short)(u >> 16);
}
__device__ __forceinline__ float bf2f(unsigned short h) {
  return __uint_as_float(((unsigned int)h) << 16);
}
__device__ __forceinline__ float fast_tanh(float x) {
  x = fminf(fmaxf(x, -20.f), 20.f);
  float e = __expf(2.f * x);
  return (e - 1.f) / (e + 1.f);
}

// ---- WMMA fragment helpers (layouts per CDNA5 ISA 7.12.2) -----------------
// A fragment 16x32 bf16: element (m,k) at p[m*ld + k]
__device__ __forceinline__ bf16x16 load_frag_a(const unsigned short* p, int ld) {
  int lane = threadIdx.x & 31;
  int m = lane & 15, hf = lane >> 4;
  const unsigned short* row = p + m * ld + hf * 8;
  FragU f;
  f.h[0] = *(const bf16x8*)(row);
  f.h[1] = *(const bf16x8*)(row + 16);
  return f.v;
}
// B fragment 32x16 bf16: element (k,n) at p[n*ld + k] -> two 16B chunks per lane (same k order as A)
__device__ __forceinline__ bf16x16 load_frag_b(const unsigned short* p, int ld) {
  int lane = threadIdx.x & 31;
  int n = lane & 15, kh = lane >> 4;
  const unsigned short* col = p + n * ld + kh * 8;
  FragU f;
  f.h[0] = *(const bf16x8*)(col);
  f.h[1] = *(const bf16x8*)(col + 16);
  return f.v;
}
__device__ __forceinline__ f32x8 wmma_bf16(bf16x16 a, bf16x16 b, f32x8 c) {
  return __builtin_amdgcn_wmma_f32_16x16x32_bf16(false, a, false, b, (short)0, c, false, false);
}
// D layout: vgpr v, lane l -> m = v + 8*(l>>4), n = l&15

// ---- small elementwise kernels --------------------------------------------
__global__ void cvt_f32_bf16(const float* __restrict__ src,
                             unsigned short* __restrict__ dst, int n) {
  for (int i = blockIdx.x * blockDim.x + threadIdx.x; i < n;
       i += gridDim.x * blockDim.x)
    dst[i] = f2bf(src[i]);
}

__global__ void qcontext_kernel(const float* __restrict__ mge,
                                const float* __restrict__ Wqc,
                                float* __restrict__ Qc) {
  int idx = blockIdx.x * blockDim.x + threadIdx.x;  // B*D threads
  int b = idx >> 9, d = idx & 511;
  float s = 0.f;
  for (int k = 0; k < DD; ++k) s = fmaf(mge[b * DD + k], Wqc[k * DD + d], s);
  Qc[idx] = s;
}

__global__ void gather_ctx(const float* __restrict__ emb,
                           const int* __restrict__ prev_a,
                           unsigned short* __restrict__ ctx) {
  // ctx[(b*T + t)*D + d] = bf16(emb[(b*N + prev_a[t,b])*D + d])
  const long total = (long)TT * BB * DD;
  for (long i = blockIdx.x * (long)blockDim.x + threadIdx.x; i < total;
       i += (long)gridDim.x * blockDim.x) {
    int d = (int)(i & 511);
    int r = (int)(i >> 9);
    int b = r >> 7, t = r & 127;
    int a = prev_a[t * BB + b];
    ctx[i] = f2bf(emb[((size_t)b * NN + a) * DD + d]);
  }
}

__global__ void q_epilogue(const unsigned short* __restrict__ qlin,
                           const float* __restrict__ Qc,
                           const float* __restrict__ cap,
                           const float* __restrict__ Wqs_full,
                           unsigned short* __restrict__ Qh) {
  const long total = (long)TT * BB * DD;
  for (long i = blockIdx.x * (long)blockDim.x + threadIdx.x; i < total;
       i += (long)gridDim.x * blockDim.x) {
    int d = (int)(i & 511);
    int r = (int)(i >> 9);
    int b = r >> 7, t = r & 127;
    float v = bf2f(qlin[i]) + Qc[b * DD + d] +
              (1.0f - cap[t * BB + b]) * Wqs_full[(size_t)512 * DD + d];
    int h = d >> 6;
    Qh[(((size_t)(b * HH + h) * TT + t) << 6) + (d & 63)] = f2bf(v);
  }
}

// ---- GEMM: C[M,512](bf16) = A[M,512](bf16) x W[512,512](bf16) --------------
// block = 256 threads (8 waves); tile M=128, N=128; K chunks of 64 with
// register prefetch of the next chunk overlapping the WMMA chain.
__global__ void gemm_bf16(const unsigned short* __restrict__ A,
                          const unsigned short* __restrict__ Wb,
                          unsigned short* __restrict__ C, int M) {
  __shared__ __attribute__((aligned(32))) unsigned short As[128 * 64];   // [m][k]
  __shared__ __attribute__((aligned(32))) unsigned short Bs[128 * 64];   // [n][k]
  int tid = threadIdx.x, w = tid >> 5;
  int m0 = blockIdx.x * 128;
  int n0 = blockIdx.y * 128;
  f32x8 acc[8];
  { f32x8 z = {}; for (int i = 0; i < 8; ++i) acc[i] = z; }

  unsigned int apre[16], bpre[16];
  #pragma unroll
  for (int i = 0; i < 16; ++i) {        // prefetch chunk k0=0
    int e = tid + 256 * i;
    int r = e >> 5, c = e & 31;         // A: 32 u32 per row of 64 bf16
    apre[i] = *(const unsigned int*)(A + (size_t)(m0 + r) * 512 + c * 2);
    int k = e >> 6, n2 = e & 63;        // B: 64 u32 (=128 bf16) per k-row
    bpre[i] = *(const unsigned int*)(Wb + (size_t)k * 512 + n0 + n2 * 2);
  }

  for (int k0 = 0; k0 < 512; k0 += 64) {
    __syncthreads();
    #pragma unroll
    for (int i = 0; i < 16; ++i) {      // commit prefetched chunk to LDS
      int e = tid + 256 * i;
      int r = e >> 5, c = e & 31;
      *(unsigned int*)(As + r * 64 + c * 2) = apre[i];
      int k = e >> 6, n2 = e & 63;
      unsigned int bv = bpre[i];        // transpose W -> Bs[n][k]
      Bs[(2 * n2) * 64 + k]     = (unsigned short)(bv & 0xFFFFu);
      Bs[(2 * n2 + 1) * 64 + k] = (unsigned short)(bv >> 16);
    }
    __syncthreads();
    if (k0 + 64 < 512) {                // prefetch next chunk (overlaps WMMAs)
      #pragma unroll
      for (int i = 0; i < 16; ++i) {
        int e = tid + 256 * i;
        int r = e >> 5, c = e & 31;
        apre[i] = *(const unsigned int*)(A + (size_t)(m0 + r) * 512 + (k0 + 64) + c * 2);
        int k = e >> 6, n2 = e & 63;
        bpre[i] = *(const unsigned int*)(Wb + (size_t)(k0 + 64 + k) * 512 + n0 + n2 * 2);
      }
    }
    #pragma unroll
    for (int kc = 0; kc < 2; ++kc) {
      bf16x16 a = load_frag_a(As + w * 16 * 64 + kc * 32, 64);
      #pragma unroll
      for (int g = 0; g < 2; ++g) {     // batch 4 B-frag loads per wait
        bf16x16 bfr[4];
        #pragma unroll
        for (int j = 0; j < 4; ++j)
          bfr[j] = load_frag_b(Bs + (g * 4 + j) * 16 * 64 + kc * 32, 64);
        #pragma unroll
        for (int j = 0; j < 4; ++j)
          acc[g * 4 + j] = wmma_bf16(a, bfr[j], acc[g * 4 + j]);
      }
    }
  }
  int lane = tid & 31, nn = lane & 15, mh = lane >> 4;
  #pragma unroll
  for (int nt = 0; nt < 8; ++nt) {
    #pragma unroll
    for (int v = 0; v < 8; ++v) {
      int m = v + 8 * mh;
      C[(size_t)(m0 + w * 16 + m) * 512 + n0 + nt * 16 + nn] = f2bf(acc[nt][v]);
    }
  }
}

// ---- fused attention per (b,h): S = Q Kt, softmax(masked), O = P V ---------
__global__ void attn_kernel(const unsigned short* __restrict__ Qh,
                            const unsigned short* __restrict__ K,
                            const unsigned short* __restrict__ V,
                            const int* __restrict__ mask,
                            unsigned short* __restrict__ mha) {
  extern __shared__ __attribute__((aligned(32))) unsigned short sm[];
  unsigned short* Qs  = sm;                  // [128][64]
  unsigned short* Ks  = sm + 8192;           // [512][64]   (n,d) d-contig
  unsigned short* VTs = sm + 8192 + 32768;   // [64][512]   (d,n) n-contig
  unsigned short* Ps  = sm + 8192 + 65536;   // 8 waves x [16][32]
  int tid = threadIdx.x, lane = tid & 31, w = tid >> 5;
  int z = blockIdx.x;
  int b = z >> 3, h = z & 7;

  {  // stage Q (contiguous)
    const unsigned int* src = (const unsigned int*)(Qh + (size_t)z * TT * 64);
    unsigned int* dst = (unsigned int*)Qs;
    #pragma unroll
    for (int i = 0; i < 16; ++i) dst[tid + 256 * i] = src[tid + 256 * i];
  }
  #pragma unroll 8
  for (int i = 0; i < 64; ++i) {  // stage K rows (u32 pairs)
    int e = tid + 256 * i;
    int r = e >> 5, c = e & 31;
    *(unsigned int*)(Ks + r * 64 + c * 2) =
        *(const unsigned int*)(K + (size_t)(b * NN + r) * 512 + h * 64 + c * 2);
  }
  #pragma unroll 8
  for (int i = 0; i < 128; ++i) {  // stage V transposed
    int e = tid + 256 * i;
    int n = e >> 6, d = e & 63;
    VTs[d * 512 + n] = V[(size_t)(b * NN + n) * 512 + h * 64 + d];
  }
  __syncthreads();

  int row0 = w * 16;
  int mh2 = lane >> 4, nn = lane & 15;
  const float scale = 0.125f;  // 1/sqrt(64)

  // Q fragments are loop-invariant: hoist into registers once
  bf16x16 aq0 = load_frag_a(Qs + row0 * 64, 64);
  bf16x16 aq1 = load_frag_a(Qs + row0 * 64 + 32, 64);

  float mi[8], li[8];
  #pragma unroll
  for (int v = 0; v < 8; ++v) { mi[v] = -1e30f; li[v] = 0.f; }

  // pass 1: online row max / sumexp
  #pragma unroll 1
  for (int nt = 0; nt < 32; ++nt) {
    bf16x16 b0 = load_frag_b(Ks + nt * 16 * 64, 64);
    bf16x16 b1 = load_frag_b(Ks + nt * 16 * 64 + 32, 64);
    f32x8 acc = {};
    acc = wmma_bf16(aq0, b0, acc);
    acc = wmma_bf16(aq1, b1, acc);
    int n_g = nt * 16 + nn;
    #pragma unroll
    for (int v = 0; v < 8; ++v) {
      int t_g = row0 + v + 8 * mh2;
      float s = acc[v] * scale;
      if (mask[((size_t)t_g * BB + b) * NN + n_g] == 1) s = NEGV;
      float mn = fmaxf(mi[v], s);
      li[v] = li[v] * __expf(mi[v] - mn) + __expf(s - mn);
      mi[v] = mn;
    }
  }
  float invl[8];
  #pragma unroll
  for (int v = 0; v < 8; ++v) {
    float m = mi[v], l = li[v];
    #pragma unroll
    for (int off = 1; off < 16; off <<= 1) {   // combine across the 16-lane half
      float mo = __shfl_xor(m, off, 32);
      float lo = __shfl_xor(l, off, 32);
      float mn = fmaxf(m, mo);
      l = l * __expf(m - mn) + lo * __expf(mo - mn);
      m = mn;
    }
    mi[v] = m;
    invl[v] = 1.0f / l;
  }

  // pass 2: recompute S, P = exp(S-m)/l, O += P V
  unsigned short* Pw = Ps + w * 512;
  f32x8 acc2[4];
  { f32x8 zz = {}; for (int dt = 0; dt < 4; ++dt) acc2[dt] = zz; }
  #pragma unroll 1
  for (int c = 0; c < 16; ++c) {
    #pragma unroll
    for (int hh = 0; hh < 2; ++hh) {
      int nt = c * 2 + hh;
      bf16x16 b0 = load_frag_b(Ks + nt * 16 * 64, 64);
      bf16x16 b1 = load_frag_b(Ks + nt * 16 * 64 + 32, 64);
      f32x8 acc = {};
      acc = wmma_bf16(aq0, b0, acc);
      acc = wmma_bf16(aq1, b1, acc);
      int n_g = nt * 16 + nn;
      #pragma unroll
      for (int v = 0; v < 8; ++v) {
        int t_g = row0 + v + 8 * mh2;
        float s = acc[v] * scale;
        if (mask[((size_t)t_g * BB + b) * NN + n_g] == 1) s = NEGV;
        float p = __expf(s - mi[v]) * invl[v];
        int m = v + 8 * mh2;
        Pw[m * 32 + hh * 16 + nn] = f2bf(p);   // per-wave LDS scratch
      }
    }
    bf16x16 a = load_frag_a(Pw, 32);           // P tile as A operand
    bf16x16 bfr[4];
    #pragma unroll
    for (int dt = 0; dt < 4; ++dt)
      bfr[dt] = load_frag_b(VTs + dt * 16 * 512 + c * 32, 512);
    #pragma unroll
    for (int dt = 0; dt < 4; ++dt)
      acc2[dt] = wmma_bf16(a, bfr[dt], acc2[dt]);
  }
  #pragma unroll
  for (int dt = 0; dt < 4; ++dt) {
    #pragma unroll
    for (int v = 0; v < 8; ++v) {
      int t_g = row0 + v + 8 * mh2;
      mha[(size_t)(b * TT + t_g) * 512 + h * 64 + dt * 16 + nn] = f2bf(acc2[dt][v]);
    }
  }
}

// ---- fused pointer head per b: comp = tanh(mhaW . Kt /sqrt(D))*10, softmax --
// N processed in chunks of 8 tiles so each A fragment feeds 8 WMMAs.
__global__ void pointer_kernel(const unsigned short* __restrict__ mhaW,
                               const unsigned short* __restrict__ KT,
                               const int* __restrict__ mask,
                               float* __restrict__ out) {
  extern __shared__ __attribute__((aligned(32))) unsigned short sm2[];  // [128][512]
  int tid = threadIdx.x, lane = tid & 31, w = tid >> 5;
  int b = blockIdx.x;
  {
    const unsigned int* src = (const unsigned int*)(mhaW + (size_t)b * TT * 512);
    unsigned int* dst = (unsigned int*)sm2;
    #pragma unroll 8
    for (int i = 0; i < 128; ++i) dst[tid + 256 * i] = src[tid + 256 * i];
  }
  __syncthreads();
  int row0 = w * 16, mh2 = lane >> 4, nn = lane & 15;
  const float scale = 0.04419417382f;  // 1/sqrt(512)
  const unsigned short* Kb = KT + (size_t)b * NN * 512;
  float mi[8], li[8];
  #pragma unroll
  for (int v = 0; v < 8; ++v) { mi[v] = -1e30f; li[v] = 0.f; }

  // pass 1
  #pragma unroll 1
  for (int ntc = 0; ntc < 4; ++ntc) {
    f32x8 acc[8];
    { f32x8 z = {}; for (int j = 0; j < 8; ++j) acc[j] = z; }
    #pragma unroll 1
    for (int kc = 0; kc < 16; ++kc) {
      bf16x16 a = load_frag_a(sm2 + row0 * 512 + kc * 32, 512);
      #pragma unroll
      for (int j = 0; j < 8; ++j) {
        bf16x16 bb = load_frag_b(Kb + (size_t)(ntc * 8 + j) * 16 * 512 + kc * 32, 512);
        acc[j] = wmma_bf16(a, bb, acc[j]);
      }
    }
    #pragma unroll
    for (int j = 0; j < 8; ++j) {
      int n_g = (ntc * 8 + j) * 16 + nn;
      #pragma unroll
      for (int v = 0; v < 8; ++v) {
        int t_g = row0 + v + 8 * mh2;
        float cmp = fast_tanh(acc[j][v] * scale) * 10.f;
        if (mask[((size_t)t_g * BB + b) * NN + n_g] == 1) cmp = NEGV;
        float mn = fmaxf(mi[v], cmp);
        li[v] = li[v] * __expf(mi[v] - mn) + __expf(cmp - mn);
        mi[v] = mn;
      }
    }
  }
  float invl[8];
  #pragma unroll
  for (int v = 0; v < 8; ++v) {
    float m = mi[v], l = li[v];
    #pragma unroll
    for (int off = 1; off < 16; off <<= 1) {
      float mo = __shfl_xor(m, off, 32);
      float lo = __shfl_xor(l, off, 32);
      float mn = fmaxf(m, mo);
      l = l * __expf(m - mn) + lo * __expf(mo - mn);
      m = mn;
    }
    mi[v] = m;
    invl[v] = 1.0f / l;
  }
  // pass 2: recompute + write probabilities
  #pragma unroll 1
  for (int ntc = 0; ntc < 4; ++ntc) {
    f32x8 acc[8];
    { f32x8 z = {}; for (int j = 0; j < 8; ++j) acc[j] = z; }
    #pragma unroll 1
    for (int kc = 0; kc < 16; ++kc) {
      bf16x16 a = load_frag_a(sm2 + row0 * 512 + kc * 32, 512);
      #pragma unroll
      for (int j = 0; j < 8; ++j) {
        bf16x16 bb = load_frag_b(Kb + (size_t)(ntc * 8 + j) * 16 * 512 + kc * 32, 512);
        acc[j] = wmma_bf16(a, bb, acc[j]);
      }
    }
    #pragma unroll
    for (int j = 0; j < 8; ++j) {
      int n_g = (ntc * 8 + j) * 16 + nn;
      #pragma unroll
      for (int v = 0; v < 8; ++v) {
        int t_g = row0 + v + 8 * mh2;
        float cmp = fast_tanh(acc[j][v] * scale) * 10.f;
        if (mask[((size_t)t_g * BB + b) * NN + n_g] == 1) cmp = NEGV;
        out[(size_t)(b * TT + t_g) * NN + n_g] = __expf(cmp - mi[v]) * invl[v];
      }
    }
  }
}

// ---- host orchestration ----------------------------------------------------
extern "C" void kernel_launch(void* const* d_in, const int* in_sizes, int n_in,
                              void* d_out, int out_size, void* d_ws, size_t ws_size,
                              hipStream_t stream) {
  const float* emb   = (const float*)d_in[0];
  const float* mge   = (const float*)d_in[1];
  const float* cap   = (const float*)d_in[2];
  const float* Wk    = (const float*)d_in[3];
  const float* WkT   = (const float*)d_in[4];
  const float* Wv    = (const float*)d_in[5];
  const float* Wqc   = (const float*)d_in[6];
  const float* Wqs   = (const float*)d_in[7];   // (513, 512)
  const float* Wout  = (const float*)d_in[8];
  const int*   prevA = (const int*)d_in[9];
  const int*   mask  = (const int*)d_in[10];
  float* out = (float*)d_out;

  // workspace layout (~640 MB)
  char* base = (char*)d_ws;
  size_t o = 0;
  auto take = [&](size_t bytes) { char* p = base + o; o += (bytes + 255) & ~(size_t)255; return p; };
  const size_t BND2 = (size_t)BB * NN * DD * 2;   // 128 MB each
  unsigned short* emb_bf = (unsigned short*)take(BND2);
  unsigned short* Kbf    = (unsigned short*)take(BND2);
  unsigned short* Vbf    = (unsigned short*)take(BND2);
  unsigned short* KTbf   = (unsigned short*)take(BND2);
  unsigned short* Wk_b   = (unsigned short*)take(512 * 512 * 2);
  unsigned short* WkT_b  = (unsigned short*)take(512 * 512 * 2);
  unsigned short* Wv_b   = (unsigned short*)take(512 * 512 * 2);
  unsigned short* Wqs_b  = (unsigned short*)take(512 * 512 * 2);
  unsigned short* Wout_b = (unsigned short*)take(512 * 512 * 2);
  float*          Qc     = (float*)take((size_t)BB * DD * 4);
  unsigned short* ctx    = (unsigned short*)take((size_t)TT * BB * DD * 2); // reused as mha
  unsigned short* qlin   = (unsigned short*)take((size_t)TT * BB * DD * 2); // reused as mhaW
  unsigned short* Qh     = (unsigned short*)take((size_t)TT * BB * DD * 2);
  unsigned short* mha  = ctx;    // ctx dead after q GEMM
  unsigned short* mhaW = qlin;   // qlin dead after q_epilogue

  // 1) inputs -> bf16 (embeddings in bf16 fit the 192 MB L2; weights tiny)
  cvt_f32_bf16<<<4096, 256, 0, stream>>>(emb, emb_bf, BB * NN * DD);
  cvt_f32_bf16<<<256, 256, 0, stream>>>(Wk,   Wk_b,   512 * 512);
  cvt_f32_bf16<<<256, 256, 0, stream>>>(WkT,  WkT_b,  512 * 512);
  cvt_f32_bf16<<<256, 256, 0, stream>>>(Wv,   Wv_b,   512 * 512);
  cvt_f32_bf16<<<256, 256, 0, stream>>>(Wqs,  Wqs_b,  512 * 512);  // first 512 rows
  cvt_f32_bf16<<<256, 256, 0, stream>>>(Wout, Wout_b, 512 * 512);

  // 2) projections: K, V, K_tanh  (M = B*N = 131072)
  dim3 gP(131072 / 128, 4);
  gemm_bf16<<<gP, 256, 0, stream>>>(emb_bf, Wk_b,  Kbf,  131072);
  gemm_bf16<<<gP, 256, 0, stream>>>(emb_bf, Wv_b,  Vbf,  131072);
  gemm_bf16<<<gP, 256, 0, stream>>>(emb_bf, WkT_b, KTbf, 131072);

  // 3) Q_context and all-step queries (scan has no carry -> batch all T steps)
  qcontext_kernel<<<512, 256, 0, stream>>>(mge, Wqc, Qc);
  gather_ctx<<<2048, 256, 0, stream>>>(emb, prevA, ctx);
  dim3 gQ(32768 / 128, 4);
  gemm_bf16<<<gQ, 256, 0, stream>>>(ctx, Wqs_b, qlin, 32768);
  q_epilogue<<<2048, 256, 0, stream>>>(qlin, Qc, cap, Wqs, Qh);

  // 4) fused attention over all (b,h): one block each, 144 KB dynamic LDS
  attn_kernel<<<BB * HH, 256, 147456, stream>>>(Qh, Kbf, Vbf, mask, mha);

  // 5) mhaW = mha @ Wout
  gemm_bf16<<<gQ, 256, 0, stream>>>(mha, Wout_b, mhaW, 32768);

  // 6) tanh-clipped pointer softmax -> out[B,T,N]; 128 KB dynamic LDS
  pointer_kernel<<<BB, 256, 131072, stream>>>(mhaW, KTbf, mask, out);

  (void)in_sizes; (void)n_in; (void)out_size; (void)ws_size;
}
